// VanillaLSTM_73529840107567
// MI455X (gfx1250) — matrix operand order repaired and checked by reading, not compiled, not run on hardware
//
#include <hip/hip_runtime.h>
#include <hip/hip_bf16.h>
#include <math.h>

typedef __bf16 bf16;
typedef __attribute__((ext_vector_type(16))) __bf16 v16bf;
typedef __attribute__((ext_vector_type(8)))  float   v8f;

#define B_TOT   4096
#define T_STEPS 64
#define CELLD   128
#define GATES   512
#define EPSF    1e-5f

union U16bf { uint4 u[2]; v16bf v; };
union Pk2   { unsigned int u; bf16 h[2]; };
union Pk8   { uint4 u; bf16 h[8]; };

__device__ __forceinline__ float sigf(float x) {
  return __builtin_amdgcn_rcpf(1.f + __expf(-x));        // v_rcp_f32 + v_exp_f32
}
__device__ __forceinline__ float tanhfast(float x) {
  return 2.f * __builtin_amdgcn_rcpf(1.f + __expf(-2.f * x)) - 1.f;
}

// ---------------------------------------------------------------------------
// K0: per-timestep raw moments of the 2-channel input over the batch dim.
// tstats[t*8 + {0..4}] = {sum x0, sum x1, sum x0^2, sum x1^2, sum x0*x1}
// ---------------------------------------------------------------------------
__global__ __launch_bounds__(256)
void input_tstats_kernel(const float* __restrict__ x, float* __restrict__ tstats)
{
  __shared__ float red[5][256];
  const int t = blockIdx.x, tid = threadIdx.x;
  float s0 = 0.f, s1 = 0.f, q0 = 0.f, q1 = 0.f, q01 = 0.f;
  for (int b = tid; b < B_TOT; b += 256) {
    const float* p = x + ((size_t)b * T_STEPS + t) * 2;
    float x0 = p[0], x1 = p[1];
    s0 += x0; s1 += x1; q0 += x0 * x0; q1 += x1 * x1; q01 += x0 * x1;
  }
  red[0][tid] = s0; red[1][tid] = s1; red[2][tid] = q0; red[3][tid] = q1; red[4][tid] = q01;
  __syncthreads();
  for (int off = 128; off > 0; off >>= 1) {
    if (tid < off)
      for (int k = 0; k < 5; ++k) red[k][tid] += red[k][tid + off];
    __syncthreads();
  }
  if (tid < 5) tstats[t * 8 + tid] = red[tid][0];
}

// ---------------------------------------------------------------------------
// K0b: whole-(B*T) input-BN stats folded with gamma/beta.
// gstats = {mu0, mu1, a0=rs0*g0, a1=rs1*g1, beta0, beta1}
// ---------------------------------------------------------------------------
__global__ __launch_bounds__(64)
void input_gstats_kernel(const float* __restrict__ tstats,
                         const float* __restrict__ in_gamma,
                         const float* __restrict__ in_beta,
                         float* __restrict__ gstats)
{
  __shared__ float red[4][64];
  const int tid = threadIdx.x;
  red[0][tid] = tstats[tid * 8 + 0];
  red[1][tid] = tstats[tid * 8 + 1];
  red[2][tid] = tstats[tid * 8 + 2];
  red[3][tid] = tstats[tid * 8 + 3];
  __syncthreads();
  for (int off = 32; off > 0; off >>= 1) {
    if (tid < off)
      for (int k = 0; k < 4; ++k) red[k][tid] += red[k][tid + off];
    __syncthreads();
  }
  if (tid == 0) {
    const float N = (float)B_TOT * (float)T_STEPS;
    float mu0 = red[0][0] / N, mu1 = red[1][0] / N;
    float v0 = red[2][0] / N - mu0 * mu0;
    float v1 = red[3][0] / N - mu1 * mu1;
    gstats[0] = mu0; gstats[1] = mu1;
    gstats[2] = rsqrtf(v0 + EPSF) * in_gamma[0];
    gstats[3] = rsqrtf(v1 + EPSF) * in_gamma[1];
    gstats[4] = in_beta[0]; gstats[5] = in_beta[1];
  }
}

// ---------------------------------------------------------------------------
// Kw: weights -> bf16, combined gate bias.
// ---------------------------------------------------------------------------
__global__ __launch_bounds__(256)
void wconv_kernel(const float* __restrict__ wih, const float* __restrict__ whh,
                  const float* __restrict__ bih, const float* __restrict__ bhh,
                  bf16* __restrict__ wih_b, bf16* __restrict__ whh_b,
                  float* __restrict__ bias)
{
  int i = blockIdx.x * blockDim.x + threadIdx.x;
  if (i < GATES * CELLD) { wih_b[i] = (bf16)wih[i]; whh_b[i] = (bf16)whh[i]; }
  if (i < GATES) bias[i] = bih[i] + bhh[i];
}

// ---------------------------------------------------------------------------
// K1: e = relu(BN_t(Linear(BN(x)))) with BN_t stats derived in closed form
// from per-t moments of the 2-dim input. Output bf16 [B][T][128].
// ---------------------------------------------------------------------------
__global__ __launch_bounds__(128)
void embed_kernel(const float* __restrict__ x, const float* __restrict__ w_emb,
                  const float* __restrict__ b_emb, const float* __restrict__ eg,
                  const float* __restrict__ ebta, const float* __restrict__ tstats,
                  const float* __restrict__ gstats, bf16* __restrict__ e)
{
  const int t = blockIdx.x;          // 0..63
  const int bch = blockIdx.y;        // chunk of 32 batch rows
  const int ch = threadIdx.x;        // 0..127
  const float mu0 = gstats[0], mu1 = gstats[1];
  const float a0 = gstats[2], a1 = gstats[3];
  const float c0 = gstats[4], c1 = gstats[5];
  const float s0 = tstats[t * 8 + 0], s1 = tstats[t * 8 + 1];
  const float q0 = tstats[t * 8 + 2], q1 = tstats[t * 8 + 3], q01 = tstats[t * 8 + 4];
  const float Bn = (float)B_TOT;
  const float mb0 = s0 / Bn, mb1 = s1 / Bn;
  const float vb0 = q0 / Bn - mb0 * mb0;
  const float vb1 = q1 / Bn - mb1 * mb1;
  const float cb  = q01 / Bn - mb0 * mb1;
  const float w0 = w_emb[ch * 2 + 0], w1 = w_emb[ch * 2 + 1], be = b_emb[ch];
  const float mn0 = (mb0 - mu0) * a0 + c0;
  const float mn1 = (mb1 - mu1) * a1 + c1;
  const float vn0 = vb0 * a0 * a0, vn1 = vb1 * a1 * a1, cn = cb * a0 * a1;
  const float me = w0 * mn0 + w1 * mn1 + be;
  const float ve = w0 * w0 * vn0 + w1 * w1 * vn1 + 2.f * w0 * w1 * cn;
  const float sc = rsqrtf(ve + EPSF) * eg[ch];
  const float sh = ebta[ch];
  for (int i = 0; i < 32; ++i) {
    int b = bch * 32 + i;
    const float* p = x + ((size_t)b * T_STEPS + t) * 2;
    float xn0 = (p[0] - mu0) * a0 + c0;
    float xn1 = (p[1] - mu1) * a1 + c1;
    float ev = (xn0 * w0 + xn1 * w1 + be - me) * sc + sh;
    ev = ev > 0.f ? ev : 0.f;
    e[((size_t)b * T_STEPS + t) * CELLD + ch] = (bf16)ev;
  }
}

// ---------------------------------------------------------------------------
// K2: the LSTM. One block = 16 batch rows for all 64 steps.
// 512 threads = 16 waves (4/SIMD -> ~256 VGPR budget); wave w owns gate
// columns [w*32, w*32+32). Both weight matrices resident as B-fragments
// (128 VGPRs). The e-stream is double-buffered in LDS: 256 threads stage
// the next step's 16x128 bf16 tile (b128 copies) while WMMAs run, so each
// tile is fetched from global exactly once per block. h tile in LDS;
// cell state c in thread registers.
// ---------------------------------------------------------------------------
__global__ __launch_bounds__(512)
void lstm_kernel(const bf16* __restrict__ e,      // [B,T,128] bf16
                 const bf16* __restrict__ wih,    // [512,128] bf16
                 const bf16* __restrict__ whh,    // [512,128] bf16
                 const float* __restrict__ bias,  // [512] b_ih+b_hh
                 bf16* __restrict__ hs,           // [B,T,128] bf16
                 float* __restrict__ hout,        // [B,128]
                 float* __restrict__ cout)        // [B,128]
{
  __shared__ bf16  h_tile[16][CELLD];
  __shared__ bf16  e_lds[2][16][CELLD];
  __shared__ float gates[16][GATES];

  const int tid  = threadIdx.x;
  const int lane = tid & 31;
  const int wv   = tid >> 5;          // 0..15
  const int b0   = blockIdx.x * 16;
  const int lm   = lane & 15;
  const int half = lane >> 4;

  // B-fragments (K x N slices of Wih^T / Whh^T), resident in VGPRs.
  // B layout (16-bit 32x16): lane col n = lm; elems 0..15 = K = half*16 + j.
  v16bf Bih[2][4], Bhh[2][4];
  for (int ns = 0; ns < 2; ++ns) {
    const int n = (wv * 2 + ns) * 16 + lm;
    for (int kt = 0; kt < 4; ++kt) {
      const int k = kt * 32 + half * 16;
      U16bf ui, uh;
      ui.u[0] = *(const uint4*)(wih + (size_t)n * CELLD + k);
      ui.u[1] = *(const uint4*)(wih + (size_t)n * CELLD + k + 8);
      uh.u[0] = *(const uint4*)(whh + (size_t)n * CELLD + k);
      uh.u[1] = *(const uint4*)(whh + (size_t)n * CELLD + k + 8);
      Bih[ns][kt] = ui.v;
      Bhh[ns][kt] = uh.v;
    }
  }
  const float bias0 = bias[(wv * 2 + 0) * 16 + lm];
  const float bias1 = bias[(wv * 2 + 1) * 16 + lm];

  for (int i = tid; i < 16 * CELLD; i += 512)
    ((bf16*)h_tile)[i] = (bf16)0.0f;

  // per-thread cell state: 2 rows x 2 adjacent channels
  float creg[4] = {0.f, 0.f, 0.f, 0.f};

  // stage e tile for timestep tn into LDS buffer nb (16 rows x 256 bytes)
  auto stage = [&](int tn, int nb) {
    if (tid < 256 && tn < T_STEPS) {
      const int row = tid >> 4, seg = tid & 15;
      uint4 v = *(const uint4*)(e + ((size_t)(b0 + row) * T_STEPS + tn) * CELLD + seg * 8);
      *(uint4*)&e_lds[nb][row][seg * 8] = v;
    }
  };

  stage(0, 0);
  __syncthreads();

  auto step = [&](int t, int buf) {
    // stage next step's tile into the other buffer; its LDS writes are
    // published by this step's second barrier, reads happen next step.
    stage(t + 1, buf ^ 1);

    v8f acc0, acc1;
    for (int r = 0; r < 8; ++r) { acc0[r] = bias0; acc1[r] = bias1; }

    // A layout (16-bit 16x32): row M = lm; elems 0..15 = K kb..kb+15,
    // kb = half*16.
    v16bf af[4];
    for (int kt = 0; kt < 4; ++kt) {
      U16bf ua;
      ua.u[0] = *(const uint4*)(&e_lds[buf][lm][kt * 32 + half * 16]);
      ua.u[1] = *(const uint4*)(&e_lds[buf][lm][kt * 32 + half * 16 + 8]);
      af[kt] = ua.v;
    }
    for (int kt = 0; kt < 4; ++kt) {
      acc0 = __builtin_amdgcn_wmma_f32_16x16x32_bf16(false, af[kt], false, Bih[0][kt],
                                                     (short)0, acc0, false, false);
      acc1 = __builtin_amdgcn_wmma_f32_16x16x32_bf16(false, af[kt], false, Bih[1][kt],
                                                     (short)0, acc1, false, false);
    }
    for (int kt = 0; kt < 4; ++kt) {
      U16bf ua;
      ua.u[0] = *(const uint4*)(&h_tile[lm][kt * 32 + half * 16]);
      ua.u[1] = *(const uint4*)(&h_tile[lm][kt * 32 + half * 16 + 8]);
      af[kt] = ua.v;
    }
    for (int kt = 0; kt < 4; ++kt) {
      acc0 = __builtin_amdgcn_wmma_f32_16x16x32_bf16(false, af[kt], false, Bhh[0][kt],
                                                     (short)0, acc0, false, false);
      acc1 = __builtin_amdgcn_wmma_f32_16x16x32_bf16(false, af[kt], false, Bhh[1][kt],
                                                     (short)0, acc1, false, false);
    }

    // C layout: VGPR r -> M = r + half*8, N = lm (within tile).
    for (int r = 0; r < 8; ++r) {
      const int m = r + half * 8;
      gates[m][(wv * 2 + 0) * 16 + lm] = acc0[r];
      gates[m][(wv * 2 + 1) * 16 + lm] = acc1[r];
    }
    __syncthreads();

    // elementwise cell update: 2 rows x 2 adjacent channels per thread,
    // h written as packed-bf16 b32 stores.
    for (int j = 0; j < 2; ++j) {
      const int bb = (tid >> 6) + 8 * j;
      const int ch = (tid & 63) * 2;
      float gi0 = gates[bb][ch],             gi1 = gates[bb][ch + 1];
      float gf0 = gates[bb][CELLD + ch],     gf1 = gates[bb][CELLD + ch + 1];
      float gg0 = gates[bb][2 * CELLD + ch], gg1 = gates[bb][2 * CELLD + ch + 1];
      float go0 = gates[bb][3 * CELLD + ch], go1 = gates[bb][3 * CELLD + ch + 1];
      float cs0 = sigf(gf0) * creg[2 * j + 0] + sigf(gi0) * tanhfast(gg0);
      float cs1 = sigf(gf1) * creg[2 * j + 1] + sigf(gi1) * tanhfast(gg1);
      float hv0 = sigf(go0) * tanhfast(cs0);
      float hv1 = sigf(go1) * tanhfast(cs1);
      creg[2 * j + 0] = cs0;
      creg[2 * j + 1] = cs1;
      Pk2 pk; pk.h[0] = (bf16)hv0; pk.h[1] = (bf16)hv1;
      *(unsigned int*)&h_tile[bb][ch] = pk.u;
      *(unsigned int*)&hs[((size_t)(b0 + bb) * T_STEPS + t) * CELLD + ch] = pk.u;
      if (t == T_STEPS - 1) {
        hout[(size_t)(b0 + bb) * CELLD + ch]     = hv0;
        hout[(size_t)(b0 + bb) * CELLD + ch + 1] = hv1;
        cout[(size_t)(b0 + bb) * CELLD + ch]     = cs0;
        cout[(size_t)(b0 + bb) * CELLD + ch + 1] = cs1;
      }
    }
    __syncthreads();
  };

#pragma unroll 1
  for (int t = 0; t < T_STEPS; t += 2) {
    step(t,     0);   // consume buffer 0, stage into buffer 1
    step(t + 1, 1);   // consume buffer 1, stage into buffer 0
  }
}

// ---------------------------------------------------------------------------
// K3a: y = hs . w_out^T + b_out, plus per-(t, o, block) partial sums of
// y and y^2 (deterministic two-stage reduction, no float atomics).
// ---------------------------------------------------------------------------
__global__ __launch_bounds__(320)
void outproj_kernel(const bf16* __restrict__ hs, const float* __restrict__ w_out,
                    const float* __restrict__ b_out, float* __restrict__ y,
                    float* __restrict__ ypart)  // [T][64 blocks][5][2]
{
  __shared__ float red[2][5][64];
  const int t = blockIdx.x, bc = blockIdx.y, tid = threadIdx.x;
  const int bl = tid / 5, o = tid % 5;
  const int b = bc * 64 + bl;
  const bf16* hrow = hs + ((size_t)b * T_STEPS + t) * CELLD;
  float acc = b_out[o];
  for (int cv = 0; cv < CELLD; cv += 8) {
    Pk8 pk;
    pk.u = *(const uint4*)(hrow + cv);
    for (int k = 0; k < 8; ++k)
      acc += (float)pk.h[k] * w_out[o * CELLD + cv + k];
  }
  y[((size_t)b * T_STEPS + t) * 5 + o] = acc;
  red[0][o][bl] = acc;
  red[1][o][bl] = acc * acc;
  __syncthreads();
  if (tid < 10) {
    const int oo = tid % 5, ss = tid / 5;
    float s = 0.f;
    for (int i = 0; i < 64; ++i) s += red[ss][oo][i];
    ypart[(((size_t)t * 64 + bc) * 5 + oo) * 2 + ss] = s;
  }
}

__global__ __launch_bounds__(320)
void outstats_kernel(const float* __restrict__ ypart, float* __restrict__ ystats)
{
  const int tid = threadIdx.x;        // 0..319 -> (t, o)
  const int t = tid / 5, o = tid % 5;
  float s = 0.f, q = 0.f;
  for (int bc = 0; bc < 64; ++bc) {
    s += ypart[(((size_t)t * 64 + bc) * 5 + o) * 2 + 0];
    q += ypart[(((size_t)t * 64 + bc) * 5 + o) * 2 + 1];
  }
  ystats[(t * 5 + o) * 2 + 0] = s;
  ystats[(t * 5 + o) * 2 + 1] = q;
}

// ---------------------------------------------------------------------------
// K3b: output batchnorm -> d_out.
// ---------------------------------------------------------------------------
__global__ __launch_bounds__(256)
void outbn_kernel(const float* __restrict__ y, const float* __restrict__ ystats,
                  const float* __restrict__ og, const float* __restrict__ ob,
                  float* __restrict__ out)
{
  const int idx = blockIdx.x * blockDim.x + threadIdx.x;
  if (idx >= B_TOT * T_STEPS * 5) return;
  const int o = idx % 5;
  const int t = (idx / 5) % T_STEPS;
  const float Bn = (float)B_TOT;
  const float s = ystats[(t * 5 + o) * 2 + 0];
  const float q = ystats[(t * 5 + o) * 2 + 1];
  const float m = s / Bn;
  const float v = q / Bn - m * m;
  out[idx] = (y[idx] - m) * rsqrtf(v + EPSF) * og[o] + ob[o];
}

// ---------------------------------------------------------------------------
// Workspace layout (bytes)
// ---------------------------------------------------------------------------
#define OFF_E      ((size_t)0)                                    // bf16 [B,T,128]
#define OFF_HS     (OFF_E  + (size_t)B_TOT * T_STEPS * CELLD * 2) // bf16 [B,T,128]
#define OFF_Y      (OFF_HS + (size_t)B_TOT * T_STEPS * CELLD * 2) // f32  [B,T,5]
#define OFF_WIH    (OFF_Y  + (size_t)B_TOT * T_STEPS * 5 * 4)     // bf16 [512,128]
#define OFF_WHH    (OFF_WIH + (size_t)GATES * CELLD * 2)
#define OFF_BIAS   (OFF_WHH + (size_t)GATES * CELLD * 2)          // f32 [512]
#define OFF_TST    (OFF_BIAS + (size_t)GATES * 4)                 // f32 [64][8]
#define OFF_GST    (OFF_TST + (size_t)64 * 8 * 4)                 // f32 [8]
#define OFF_YPART  (OFF_GST + (size_t)64)                         // f32 [64][64][5][2]
#define OFF_YST    (OFF_YPART + (size_t)64 * 64 * 5 * 2 * 4)      // f32 [64][5][2]

extern "C" void kernel_launch(void* const* d_in, const int* in_sizes, int n_in,
                              void* d_out, int out_size, void* d_ws, size_t ws_size,
                              hipStream_t stream) {
  const float* x     = (const float*)d_in[0];
  const float* in_g  = (const float*)d_in[1];
  const float* in_b  = (const float*)d_in[2];
  const float* w_emb = (const float*)d_in[3];
  const float* b_emb = (const float*)d_in[4];
  const float* emb_g = (const float*)d_in[5];
  const float* emb_b = (const float*)d_in[6];
  const float* w_ih  = (const float*)d_in[7];
  const float* b_ih  = (const float*)d_in[8];
  const float* w_hh  = (const float*)d_in[9];
  const float* b_hh  = (const float*)d_in[10];
  const float* w_out = (const float*)d_in[11];
  const float* b_out = (const float*)d_in[12];
  const float* out_g = (const float*)d_in[13];
  const float* out_b = (const float*)d_in[14];

  char* ws = (char*)d_ws;
  bf16*  e_bf   = (bf16*) (ws + OFF_E);
  bf16*  hs_bf  = (bf16*) (ws + OFF_HS);
  float* y      = (float*)(ws + OFF_Y);
  bf16*  wih_b  = (bf16*) (ws + OFF_WIH);
  bf16*  whh_b  = (bf16*) (ws + OFF_WHH);
  float* bias   = (float*)(ws + OFF_BIAS);
  float* tstats = (float*)(ws + OFF_TST);
  float* gstats = (float*)(ws + OFF_GST);
  float* ypart  = (float*)(ws + OFF_YPART);
  float* ystats = (float*)(ws + OFF_YST);

  float* out  = (float*)d_out;                       // [B,T,5]
  float* hout = out + (size_t)B_TOT * T_STEPS * 5;   // [B,128]
  float* cout = hout + (size_t)B_TOT * CELLD;        // [B,128]

  wconv_kernel<<<dim3((GATES * CELLD + 255) / 256), 256, 0, stream>>>(
      w_ih, w_hh, b_ih, b_hh, wih_b, whh_b, bias);
  input_tstats_kernel<<<dim3(T_STEPS), 256, 0, stream>>>(x, tstats);
  input_gstats_kernel<<<dim3(1), 64, 0, stream>>>(tstats, in_g, in_b, gstats);
  embed_kernel<<<dim3(T_STEPS, B_TOT / 32), 128, 0, stream>>>(
      x, w_emb, b_emb, emb_g, emb_b, tstats, gstats, e_bf);
  lstm_kernel<<<dim3(B_TOT / 16), 512, 0, stream>>>(
      e_bf, wih_b, whh_b, bias, hs_bf, hout, cout);
  outproj_kernel<<<dim3(T_STEPS, B_TOT / 64), 320, 0, stream>>>(
      hs_bf, w_out, b_out, y, ypart);
  outstats_kernel<<<dim3(1), 320, 0, stream>>>(ypart, ystats);
  outbn_kernel<<<dim3((B_TOT * T_STEPS * 5 + 255) / 256), 256, 0, stream>>>(
      y, ystats, out_g, out_b, out);
}
